// MimiEuclideanCodebook_89455578841699
// MI455X (gfx1250) — matrix operand order repaired and checked by reading, not compiled, not run on hardware
//
#include <hip/hip_runtime.h>

// MimiEuclideanCodebook for MI455X (gfx1250): bf16x3-compensated WMMA distance
// GEMM with fused register argmin + EMA scatter. B tiles are staged into LDS
// with gfx1250 async global->LDS copies (ASYNCcnt) and double-buffered so the
// copy of tile kt+1 overlaps the 24 WMMAs of tile kt.

#define DDIM      256
#define KCODES    2048
#define EPSV      1e-5f
#define DECAYV    0.99f
#define NTILES    (KCODES / 16)

typedef __attribute__((ext_vector_type(16))) __bf16 v16bf;
typedef __attribute__((ext_vector_type(8)))  float  v8f;

union Frag { unsigned u[8]; v16bf v; };

static __device__ inline unsigned short f2bf(float f) {
  unsigned u = __builtin_bit_cast(unsigned, f);
  u += 0x7FFFu + ((u >> 16) & 1u);            // round-to-nearest-even
  return (unsigned short)(u >> 16);
}
static __device__ inline float bf2f(unsigned short b) {
  return __builtin_bit_cast(float, (unsigned)b << 16);
}
static __device__ inline unsigned packbf(unsigned short lo, unsigned short hi) {
  return (unsigned)lo | ((unsigned)hi << 16);
}

// Async 16B global->LDS copy (GVS form: saddr pair + 32-bit vgpr byte offset).
// Tracked by ASYNCcnt; drained with s_wait_asynccnt.
static __device__ inline void async_copy_b128(unsigned ldsByteAddr,
                                              unsigned long long gBase,
                                              unsigned gByteOff) {
  asm volatile("global_load_async_to_lds_b128 %0, %1, %2"
               :: "v"(ldsByteAddr), "v"(gByteOff), "s"(gBase)
               : "memory");
}
static __device__ inline void wait_async0() {
  asm volatile("s_wait_asynccnt 0x0" ::: "memory");
}

// ---------------------------------------------------------------- zero scratch
__global__ void vq_zero_kernel(float* __restrict__ p, int n) {
  int i = blockIdx.x * blockDim.x + threadIdx.x;
  if (i < n) p[i] = 0.0f;
}

// ------------------------------------------------- prep: embed, e2, bf16 hi/lo
// Packs embed into the WMMA B-fragment dword layout:
//   slot = ((kTile*8 + dStep)*8 + j)*32 + (halfK*16 + (k&15))
// where element (k, d): dStep=d>>5, halfK=((d&31)>>3)&1, j=(((d&31)>>4)<<2)|((d&7)>>1).
__global__ __launch_bounds__(256)
void vq_prep_kernel(const float* __restrict__ embed_sum,
                    const float* __restrict__ usage,
                    float* __restrict__ embedf,
                    unsigned* __restrict__ ebH,
                    unsigned* __restrict__ ebL,
                    float* __restrict__ e2w) {
  int lane = threadIdx.x & 31;
  int wave = threadIdx.x >> 5;
  int k = blockIdx.x * 8 + wave;                 // one wave per codebook row
  float u = fmaxf(usage[k], EPSV);
  const float* es = embed_sum + (size_t)k * DDIM;
  float* ef = embedf + (size_t)k * DDIM;
  int kTile = k >> 4, np = k & 15;
  float ss = 0.0f;
  #pragma unroll
  for (int p = 0; p < 4; ++p) {                  // lane covers d = 8*lane..8*lane+7
    int d = lane * 8 + p * 2;
    float e0 = es[d] / u;
    float e1 = es[d + 1] / u;
    ef[d] = e0; ef[d + 1] = e1;
    ss += e0 * e0 + e1 * e1;
    unsigned short h0 = f2bf(e0), h1 = f2bf(e1);
    int ds = d >> 5, r = d & 31, hf = (r >> 3) & 1, j = ((r >> 4) << 2) | ((r & 7) >> 1);
    int slot = ((kTile * 8 + ds) * 8 + j) * 32 + hf * 16 + np;
    ebH[slot] = packbf(h0, h1);
    ebL[slot] = packbf(f2bf(e0 - bf2f(h0)), f2bf(e1 - bf2f(h1)));
  }
  #pragma unroll
  for (int off = 16; off >= 1; off >>= 1) ss += __shfl_xor(ss, off, 32);
  if (lane == 0) e2w[k] = ss;
}

// ------------------------------------------------------------------ main fused
// Block = 256 threads = 8 waves; each wave owns 16 rows. B tiles double-
// buffered in LDS via async global->LDS (16KB per buffer: H | L).
__global__ __launch_bounds__(256)
void vq_main_kernel(const float* __restrict__ hidden,
                    const float* __restrict__ embedf,
                    const unsigned* __restrict__ ebH,
                    const unsigned* __restrict__ ebL,
                    const float* __restrict__ e2w,
                    float* __restrict__ out_q,
                    float* __restrict__ out_ind,
                    float* __restrict__ counts,
                    float* __restrict__ sums) {
  __shared__ unsigned lds[8192];   // 2 buffers x (2048 H + 2048 L) dwords = 32KB

  int tid = threadIdx.x;
  int lane = tid & 31;
  int wave = tid >> 5;
  int half = lane >> 4;
  int ln = lane & 15;
  int rowBase = (blockIdx.x * 8 + wave) * 16;

  // Async staging setup. Flat shared addresses keep the LDS byte offset in
  // addr[31:0] (ISA aperture rule), so truncation yields the DS address.
  unsigned long long gHb = (unsigned long long)(size_t)ebH;
  unsigned long long gLb = (unsigned long long)(size_t)ebL;
  unsigned ldsBase = (unsigned)(size_t)(void*)lds;
  unsigned tOff = (unsigned)tid * 32u;           // this thread's 32B slice

  // Issue tile 0 copy into buffer 0 (overlaps A-fragment register loads).
  {
    unsigned lH = ldsBase + tOff;
    unsigned lL = ldsBase + 8192u + tOff;
    async_copy_b128(lH,       gHb, tOff);
    async_copy_b128(lH + 16u, gHb, tOff + 16u);
    async_copy_b128(lL,       gLb, tOff);
    async_copy_b128(lL + 16u, gLb, tOff + 16u);
  }

  // A fragments (bf16 hi + lo) for this wave's 16x256 tile, kept in registers.
  // 16-bit A 16x32 layout: lane holds M=lane&15; vgpr j holds
  // K = (j>>2)*16 + half*8 + (j&3)*2 (+1).
  Frag aH[8], aL[8];
  {
    const float* xrow = hidden + (size_t)(rowBase + ln) * DDIM;
    #pragma unroll
    for (int ds = 0; ds < 8; ++ds) {
      #pragma unroll
      for (int j = 0; j < 8; ++j) {
        int d = ds * 32 + (j >> 2) * 16 + half * 8 + (j & 3) * 2;
        float f0 = xrow[d], f1 = xrow[d + 1];
        unsigned short h0 = f2bf(f0), h1 = f2bf(f1);
        aH[ds].u[j] = packbf(h0, h1);
        aL[ds].u[j] = packbf(f2bf(f0 - bf2f(h0)), f2bf(f1 - bf2f(h1)));
      }
    }
  }

  float bestVal[8];
  int   bestIdx[8];
  #pragma unroll
  for (int r = 0; r < 8; ++r) { bestVal[r] = -3.4e38f; bestIdx[r] = 0; }

  for (int kt = 0; kt < NTILES; ++kt) {
    wait_async0();            // own copies for tile kt complete (in order)
    __syncthreads();          // all waves' slices of tile kt visible

    // Prefetch tile kt+1 into the other buffer; overlaps the WMMAs below.
    // Safe: that buffer's last readers passed the end-of-iteration barrier
    // of tile kt-1.
    if (kt + 1 < NTILES) {
      unsigned bb = ((unsigned)(kt + 1) & 1u) * 16384u;
      unsigned gOff = (unsigned)(kt + 1) * 8192u + tOff;
      unsigned lH = ldsBase + bb + tOff;
      unsigned lL = ldsBase + bb + 8192u + tOff;
      async_copy_b128(lH,       gHb, gOff);
      async_copy_b128(lH + 16u, gHb, gOff + 16u);
      async_copy_b128(lL,       gLb, gOff);
      async_copy_b128(lL + 16u, gLb, gOff + 16u);
    }

    // acc = x.e - e2/2 ; maximize acc == minimize ||x-e||^2
    int bufBase = (kt & 1) ? 4096 : 0;
    float e2v = e2w[kt * 16 + ln];
    v8f acc;
    #pragma unroll
    for (int i = 0; i < 8; ++i) acc[i] = -0.5f * e2v;

    #pragma unroll
    for (int ds = 0; ds < 8; ++ds) {
      Frag bH, bL;
      #pragma unroll
      for (int j = 0; j < 8; ++j) {              // conflict-free: index = base+lane
        bH.u[j] = lds[bufBase + (ds * 8 + j) * 32 + lane];
        bL.u[j] = lds[bufBase + 2048 + (ds * 8 + j) * 32 + lane];
      }
      acc = __builtin_amdgcn_wmma_f32_16x16x32_bf16(false, aH[ds].v, false, bH.v, (short)0, acc, false, false);
      acc = __builtin_amdgcn_wmma_f32_16x16x32_bf16(false, aH[ds].v, false, bL.v, (short)0, acc, false, false);
      acc = __builtin_amdgcn_wmma_f32_16x16x32_bf16(false, aL[ds].v, false, bH.v, (short)0, acc, false, false);
    }

    int col = kt * 16 + ln;
    #pragma unroll
    for (int r = 0; r < 8; ++r) {
      if (acc[r] > bestVal[r]) { bestVal[r] = acc[r]; bestIdx[r] = col; }
    }

    __syncthreads();          // everyone done reading buf before next overwrite
  }

  // Cross-lane argmax within each 16-lane half (C layout: N = lane&15;
  // lanes 0-15 hold M=r, lanes 16-31 hold M=r+8). Ties -> smaller index
  // (matches argmin first-occurrence).
  #pragma unroll
  for (int r = 0; r < 8; ++r) {
    float bv = bestVal[r]; int bi = bestIdx[r];
    #pragma unroll
    for (int m = 1; m <= 8; m <<= 1) {
      float ov = __shfl_xor(bv, m, 32);
      int   oi = __shfl_xor(bi, m, 32);
      if (ov > bv || (ov == bv && oi < bi)) { bv = ov; bi = oi; }
    }
    bestIdx[r] = bi;
    if (ln == r) out_ind[rowBase + r + half * 8] = (float)bi;
  }

  // Epilogue: quantize gather + EMA scatter (raw sums; decay applied later).
  #pragma unroll
  for (int m = 0; m < 16; ++m) {
    int idx = __shfl(bestIdx[m & 7], (m & 8) ? 16 : 0, 32);
    int row = rowBase + m;
    const float* er = embedf + (size_t)idx * DDIM;
    const float* xr = hidden + (size_t)row * DDIM;
    float* qr = out_q + (size_t)row * DDIM;
    float* sr = sums + (size_t)idx * DDIM;
    #pragma unroll
    for (int i = 0; i < 8; ++i) {
      int c = lane + i * 32;
      qr[c] = er[c];
      atomicAdd(&sr[c], xr[c]);
    }
    if (lane == 0) atomicAdd(&counts[idx], 1.0f);
  }
}

// ------------------------------------------------------------------- EMA blend
__global__ void vq_final_kernel(const float* __restrict__ usage,
                                const float* __restrict__ embed_sum,
                                const float* __restrict__ counts,
                                const float* __restrict__ sums,
                                float* __restrict__ out_usage,
                                float* __restrict__ out_es) {
  int i = blockIdx.x * blockDim.x + threadIdx.x;
  if (i < KCODES) out_usage[i] = usage[i] * DECAYV + counts[i] * (1.0f - DECAYV);
  if (i < KCODES * DDIM) out_es[i] = embed_sum[i] * DECAYV + sums[i] * (1.0f - DECAYV);
}

// ------------------------------------------------------------------- launcher
extern "C" void kernel_launch(void* const* d_in, const int* in_sizes, int n_in,
                              void* d_out, int out_size, void* d_ws, size_t ws_size,
                              hipStream_t stream) {
  const float* hidden    = (const float*)d_in[0];   // [16,4096,256]
  const float* embed_sum = (const float*)d_in[1];   // [2048,256]
  const float* usage     = (const float*)d_in[2];   // [2048]

  float* out = (float*)d_out;
  float* out_q     = out;                                   // 16*4096*256
  float* out_ind   = out + (size_t)16 * 4096 * 256;         // 16*4096
  float* out_usage = out_ind + 16 * 4096;                   // 2048
  float* out_es    = out_usage + KCODES;                    // 2048*256

  float* ws      = (float*)d_ws;
  float* embedf  = ws;                                      // 524288 f32
  float* e2w     = embedf + (size_t)KCODES * DDIM;          // 2048
  float* counts  = e2w + KCODES;                            // 2048
  float* sums    = counts + KCODES;                         // 524288
  unsigned* ebH  = (unsigned*)(sums + (size_t)KCODES * DDIM); // 262144 dwords
  unsigned* ebL  = ebH + (size_t)KCODES * DDIM / 2;           // 262144 dwords

  int nz = KCODES + KCODES * DDIM;                          // counts + sums
  vq_zero_kernel <<<(nz + 255) / 256, 256, 0, stream>>>(counts, nz);
  vq_prep_kernel <<<KCODES / 8, 256, 0, stream>>>(embed_sum, usage, embedf, ebH, ebL, e2w);
  vq_main_kernel <<<(16 * 4096) / 128, 256, 0, stream>>>(hidden, embedf, ebH, ebL, e2w,
                                                         out_q, out_ind, counts, sums);
  vq_final_kernel<<<(KCODES * DDIM) / 256, 256, 0, stream>>>(usage, embed_sum, counts, sums,
                                                             out_usage, out_es);
}
